// GATCriticWithLaser_29188597743648
// MI455X (gfx1250) — matrix operand order repaired and checked by reading, not compiled, not run on hardware
//
#include <hip/hip_runtime.h>
#include <hip/hip_bf16.h>

// ---------------------------------------------------------------------------
// GAT critic with laser CNN front-end, CDNA5 (gfx1250) implementation.
// Dense node GEMMs use V_WMMA_F32_16X16X32_F16 (f16 in, f32 accumulate),
// with the A-strip staged into LDS by the Tensor Data Mover
// (tensor_load_to_lds + s_wait_tensorcnt).  Graph message passing uses f32
// global atomics (segment softmax + scatter).
// ---------------------------------------------------------------------------

#define N_NODES 50000
#define N_EDGES 800000
#define N_BATCH 64
#define HEADS 8
#define HID 32
#define DOUT 256            // HEADS*HID
#define NEG_SLOPE 0.2f

typedef __attribute__((ext_vector_type(16))) _Float16 v16h;
typedef __attribute__((ext_vector_type(8)))  _Float16 v8h;
typedef __attribute__((ext_vector_type(8)))  float    v8f;
typedef __attribute__((ext_vector_type(4)))  unsigned int u32x4;
typedef __attribute__((ext_vector_type(8)))  int i32x8;
typedef __attribute__((ext_vector_type(4)))  int i32x4;

// ---------------------------------------------------------------------------
// Generic fill
// ---------------------------------------------------------------------------
__global__ void fill_kernel(float* p, float v, size_t n) {
    size_t i = (size_t)blockIdx.x * blockDim.x + threadIdx.x;
    if (i < n) p[i] = v;
}

// ---------------------------------------------------------------------------
// CNN front-end: one wave (32 lanes) per node.
// lidar(20) -> conv1d(16,k5,SAME)+relu -> maxpool2 -> conv1d(32,k3,SAME)+relu
// -> mean -> linear(32->16)+relu.  Output xg(25) written as f16 padded to 32.
// ---------------------------------------------------------------------------
__global__ __launch_bounds__(32)
void cnn_kernel(const float* __restrict__ x,
                const float* __restrict__ w1, const float* __restrict__ b1,
                const float* __restrict__ w2, const float* __restrict__ b2,
                const float* __restrict__ wl, const float* __restrict__ bl,
                _Float16* __restrict__ Xg) {
    __shared__ float pooled[16 * 10];
    __shared__ float mchan[32];
    const int node = blockIdx.x;
    const int l = threadIdx.x;
    const float* xn = x + (size_t)node * 29;

    if (l < 16) {
        for (int q = 0; q < 10; ++q) {
            float mx = -1e30f;
            for (int t = 0; t < 2; ++t) {
                int p = 2 * q + t;
                float acc = b1[l];
                for (int k = 0; k < 5; ++k) {
                    int idx = p + k - 2;
                    float v = (idx >= 0 && idx < 20) ? xn[idx] : 0.f;
                    acc += w1[l * 5 + k] * v;
                }
                acc = fmaxf(acc, 0.f);
                mx = fmaxf(mx, acc);
            }
            pooled[l * 10 + q] = mx;
        }
    }
    __syncthreads();
    {
        float acc = 0.f;
        for (int p = 0; p < 10; ++p) {
            float o = b2[l];
            for (int ci = 0; ci < 16; ++ci) {
                const float* wp = w2 + (size_t)(l * 16 + ci) * 3;
                for (int k = 0; k < 3; ++k) {
                    int idx = p + k - 1;
                    float v = (idx >= 0 && idx < 10) ? pooled[ci * 10 + idx] : 0.f;
                    o += wp[k] * v;
                }
            }
            acc += fmaxf(o, 0.f);
        }
        mchan[l] = acc * 0.1f;     // mean over length 10
    }
    __syncthreads();
    _Float16* out = Xg + (size_t)node * 32;
    if (l < 16) {
        float e = bl[l];
        for (int c = 0; c < 32; ++c) e += mchan[c] * wl[c * 16 + l];
        out[6 + l] = (_Float16)fmaxf(e, 0.f);
    }
    if (l < 6)             out[l]      = (_Float16)xn[20 + l];       // non_lidar
    if (l >= 6 && l < 9)   out[16 + l] = (_Float16)xn[20 + l];       // last_action -> xg[22..24]
    if (l < 7)             out[25 + l] = (_Float16)0.f;              // pad 25..31
}

// ---------------------------------------------------------------------------
// Degrees & self-loop attribute
// ---------------------------------------------------------------------------
__global__ void deg_kernel(const int* __restrict__ ei, const float* __restrict__ ea,
                           float* deg, float* slsum) {
    int e = blockIdx.x * blockDim.x + threadIdx.x;
    if (e >= N_EDGES) return;
    int d = ei[N_EDGES + e];
    atomicAdd(&deg[d], 1.f);
    atomicAdd(&slsum[d], ea[e]);
}
__global__ void slattr_kernel(float* slattr, const float* slsum, const float* deg) {
    int i = blockIdx.x * blockDim.x + threadIdx.x;
    if (i < N_NODES) slattr[i] = slsum[i] / fmaxf(deg[i], 1.f);
}

// ---------------------------------------------------------------------------
// Weight convert: W (K x 256) f32 -> WT (256 x Kpad) f16 (transposed, padded)
// ---------------------------------------------------------------------------
__global__ void convert_wt(const float* __restrict__ W, _Float16* __restrict__ WT,
                           int K, int Kpad) {
    int i = blockIdx.x * blockDim.x + threadIdx.x;
    if (i >= 256 * Kpad) return;
    int nrow = i / Kpad, k = i % Kpad;
    WT[i] = (_Float16)((k < K) ? W[(size_t)k * 256 + nrow] : 0.f);
}

// ---------------------------------------------------------------------------
// TDM helper: issue TENSOR_LOAD_TO_LDS of a 2D tile (tile_dim1 rows x
// tile_dim0 elements of 2 bytes) starting at `gaddr`, row stride = `stride0`
// elements, into LDS byte offset `lds_off`.  D# bit layout per ISA ch.8.
// This toolchain exposes the 6-arg builtin:
//   (uint32x4 g0, int32x8 g1, int32x4 g2, int32x4 g3, int32x8 gx, i32 cpol)
// ---------------------------------------------------------------------------
#if __has_builtin(__builtin_amdgcn_tensor_load_to_lds)
#define HAVE_TDM 1
__device__ __forceinline__ void tdm_load_2d_f16(unsigned int lds_off,
                                                unsigned long long gaddr,
                                                unsigned int tile_d0,
                                                unsigned int tile_d1,
                                                unsigned int stride0) {
    // group 0: count=1 | lds_addr | global_addr | type=2
    u32x4 g0;
    g0[0] = 1u;                                         // count=1, user desc
    g0[1] = lds_off;                                    // LDS byte address
    g0[2] = (unsigned int)(gaddr & 0xFFFFFFFFull);      // global_addr[31:0]
    g0[3] = (unsigned int)((gaddr >> 32) & 0x1FFFFFFull) | (2u << 30); // [56:32] | type=2
    // group 1: data_size=1 (2B), tensor dims, tile dims, strides
    unsigned int tdim0 = tile_d0;        // tensor width  (tile fully in-bounds)
    unsigned int tdim1 = tile_d1;        // tensor height
    unsigned long long s0 = stride0;     // dim0 stride (elements)
    i32x8 g1;
    g1[0] = (int)(1u << 16);                                          // wg_mask=0, data_size=1
    g1[1] = (int)((tdim0 & 0xFFFFu) << 16);                           // abar=0 | tensor_dim0 lo
    g1[2] = (int)((tdim0 >> 16) | ((tdim1 & 0xFFFFu) << 16));         // dim0 hi | dim1 lo
    g1[3] = (int)(((tdim1 >> 16) & 0xFFFFu) | (tile_d0 << 16));       // dim1 hi | tile_dim0
    g1[4] = (int)(tile_d1 & 0xFFFFu);                                 // tile_dim1 | tile_dim2=0
    g1[5] = (int)(s0 & 0xFFFFFFFFull);                                // dim0_stride lo
    g1[6] = (int)((s0 >> 32) & 0xFFFFull);                            // dim0_stride hi | dim1_stride lo=0
    g1[7] = 0;                                                        // dim1_stride hi
    // groups 2/3: higher dims = 1 so nothing is considered out-of-bounds
    i32x4 g2; g2[0] = 1; g2[1] = 1; g2[2] = 0; g2[3] = 0;             // tensor_dim2=1, dim3=1
    i32x4 g3; g3[0] = 0; g3[1] = (int)(1u << 16); g3[2] = 0; g3[3] = 0; // tensor_dim4=1
    i32x8 gx = {};                                                    // unused trailing group
    __builtin_amdgcn_tensor_load_to_lds(g0, g1, g2, g3, gx, 0);
}
#else
#define HAVE_TDM 0
#endif

// ---------------------------------------------------------------------------
// WMMA GEMM: C(M x 256) = A(M x K, f16) * B(K x 256, f16 stored transposed
// as WT(256 x K)).  Block = 256 threads = 8 waves; each wave computes TWO
// 16x16 tiles (reusing its A fragment across 2 WMMAs); block covers a
// 16-row x 256-col strip.  The 16xK A strip is DMA'd into LDS once by the
// Tensor Data Mover, then all K-steps run barrier-free out of LDS.
// M (=50000) is a multiple of 16 so no row guard is needed.
// Fragment layouts per CDNA5 ISA 7.12.2:
//   A: lane L -> M=L%16; elems 0..7 = K(+0/8 by lane-half)+j, elems 8..15 = K+16(+0/8)+j
//   B: lane L -> N=L%16; elems 0..7 = K(+0/8 by lane-half)+j, elems 8..15 = K+16(+0/8)+j
//   C: VGPR r, lane L -> M = r + (L>=16)*8, N = L%16
// ---------------------------------------------------------------------------
template <int K>
__global__ __launch_bounds__(256)
void gemm_wmma(const _Float16* __restrict__ A,
               const _Float16* __restrict__ WT,
               float* __restrict__ C) {
    __shared__ alignas(16) _Float16 As[16 * K];
    const int mtile = blockIdx.x;
    const int wave  = threadIdx.x >> 5;   // 0..7 -> pair of N tiles
    const int lane  = threadIdx.x & 31;
    const int row0  = mtile * 16;
    const int m     = lane & 15;
    const int hiA   = (lane >> 4) * 8;    // 0 or 8
    const int hiB   = (lane >> 4) * 16;   // 0 or 16

#if HAVE_TDM
    if (wave == 0) {
        unsigned int lds_off = (unsigned int)(size_t)(&As[0]);  // low 32b = LDS byte offset
        unsigned long long gaddr =
            (unsigned long long)(size_t)(A + (size_t)row0 * K);
        tdm_load_2d_f16(lds_off, gaddr, (unsigned)K, 16u, (unsigned)K);
        __builtin_amdgcn_s_wait_tensorcnt(0);
    }
    __syncthreads();
#else
    for (int t = threadIdx.x; t < 16 * K; t += 256) {
        int r = t / K, k = t % K;
        As[t] = A[(size_t)(row0 + r) * K + k];
    }
    __syncthreads();
#endif

    const int nbase0 = wave * 32 + (lane & 15);
    const int nbase1 = nbase0 + 16;
    v8f acc0 = {}, acc1 = {};
    #pragma unroll
    for (int kk = 0; kk < K; kk += 32) {
        v16h a, b0, b1;
        const _Float16* ap = &As[m * K + kk];
        v8h a_lo = *(const v8h*)(ap + hiA);
        v8h a_hi = *(const v8h*)(ap + 16 + hiA);
        const _Float16* bp0 = WT + (size_t)nbase0 * K + kk + hiA;
        const _Float16* bp1 = WT + (size_t)nbase1 * K + kk + hiA;
        v8h b0_lo = *(const v8h*)(bp0);
        v8h b0_hi = *(const v8h*)(bp0 + 16);
        v8h b1_lo = *(const v8h*)(bp1);
        v8h b1_hi = *(const v8h*)(bp1 + 16);
        #pragma unroll
        for (int j = 0; j < 8; ++j) {
            a[j]  = a_lo[j];  a[j + 8]  = a_hi[j];
            b0[j] = b0_lo[j]; b0[j + 8] = b0_hi[j];
            b1[j] = b1_lo[j]; b1[j + 8] = b1_hi[j];
        }
        acc0 = __builtin_amdgcn_wmma_f32_16x16x32_f16(
            false, a, false, b0, (short)0, acc0, false, false);
        acc1 = __builtin_amdgcn_wmma_f32_16x16x32_f16(
            false, a, false, b1, (short)0, acc1, false, false);
    }
    const int mout = row0 + (lane >> 4) * 8;
    #pragma unroll
    for (int r = 0; r < 8; ++r) {
        C[(size_t)(mout + r) * 256 + nbase0] = acc0[r];
        C[(size_t)(mout + r) * 256 + nbase1] = acc1[r];
    }
}

// ---------------------------------------------------------------------------
// Per-node attention scores: a_src[n,h], a_dst[n,h]
// ---------------------------------------------------------------------------
__global__ void att_scores(const float* __restrict__ H,
                           const float* __restrict__ att_src,
                           const float* __restrict__ att_dst,
                           float* asrc, float* adst) {
    int i = blockIdx.x * blockDim.x + threadIdx.x;
    if (i >= N_NODES * HEADS) return;
    int node = i >> 3, h = i & 7;
    const float* hp = H + (size_t)node * 256 + h * 32;
    float s1 = 0.f, s2 = 0.f;
    for (int c = 0; c < 32; ++c) {
        float v = hp[c];
        s1 += v * att_src[h * 32 + c];
        s2 += v * att_dst[h * 32 + c];
    }
    asrc[i] = s1; adst[i] = s2;
}

// c_e[h] = sum_c We[h*32+c] * att_e[h,c]   (edge feature is scalar, so the
// per-edge attention contribution collapses to eattr[e] * c_e[h])
__global__ void ce_kernel(const float* __restrict__ We,
                          const float* __restrict__ att_e, float* ce) {
    int h = threadIdx.x;
    if (h < HEADS) {
        float s = 0.f;
        for (int c = 0; c < 32; ++c) s += We[h * 32 + c] * att_e[h * 32 + c];
        ce[h] = s;
    }
}

__device__ __forceinline__ void atomicMaxF(float* addr, float val) {
    unsigned int* ua = (unsigned int*)addr;
    unsigned int old = *ua;
    while (true) {
        if (__uint_as_float(old) >= val) break;
        unsigned int assumed = old;
        old = atomicCAS(ua, assumed, __float_as_uint(val));
        if (old == assumed) break;
    }
}

// ---------------------------------------------------------------------------
// Edge softmax, 3 passes over (E + N self-loop) edges x HEADS
// ---------------------------------------------------------------------------
__global__ void edge_logit_max(const int* __restrict__ ei,
                               const float* __restrict__ ea,
                               const float* __restrict__ slattr,
                               const float* __restrict__ asrc,
                               const float* __restrict__ adst,
                               const float* __restrict__ ce,
                               float* __restrict__ logit, float* mbuf) {
    const int ES = N_EDGES + N_NODES;
    int idx = blockIdx.x * blockDim.x + threadIdx.x;
    if (idx >= ES * HEADS) return;
    int e = idx >> 3, h = idx & 7;
    int s, d; float attr;
    if (e < N_EDGES) { s = ei[e]; d = ei[N_EDGES + e]; attr = ea[e]; }
    else             { s = d = e - N_EDGES;            attr = slattr[s]; }
    float z = asrc[s * 8 + h] + adst[d * 8 + h] + attr * ce[h];
    z = (z > 0.f) ? z : NEG_SLOPE * z;       // leaky relu
    logit[idx] = z;
    atomicMaxF(&mbuf[d * 8 + h], z);
}

__global__ void edge_exp_sum(const int* __restrict__ ei,
                             float* __restrict__ logit /* in logit, out exp */,
                             const float* __restrict__ mbuf, float* sbuf) {
    const int ES = N_EDGES + N_NODES;
    int idx = blockIdx.x * blockDim.x + threadIdx.x;
    if (idx >= ES * HEADS) return;
    int e = idx >> 3, h = idx & 7;
    int d = (e < N_EDGES) ? ei[N_EDGES + e] : e - N_EDGES;
    float ex = __expf(logit[idx] - mbuf[d * 8 + h]);
    logit[idx] = ex;
    atomicAdd(&sbuf[d * 8 + h], ex);
}

__global__ void edge_message(const int* __restrict__ ei,
                             const float* __restrict__ ex,
                             const float* __restrict__ sbuf,
                             const float* __restrict__ H,
                             float* __restrict__ out) {
    const int ES = N_EDGES + N_NODES;
    int idx = blockIdx.x * blockDim.x + threadIdx.x;
    if (idx >= ES * HEADS) return;
    int e = idx >> 3, h = idx & 7;
    int s, d;
    if (e < N_EDGES) { s = ei[e]; d = ei[N_EDGES + e]; }
    else             { s = d = e - N_EDGES; }
    float w = ex[idx] / (sbuf[d * 8 + h] + 1e-16f);
    const float4* hp = (const float4*)(H + (size_t)s * 256 + h * 32);
    float* op = out + (size_t)d * 256 + h * 32;
    #pragma unroll
    for (int v = 0; v < 8; ++v) {
        float4 q = hp[v];
        atomicAdd(op + v * 4 + 0, q.x * w);
        atomicAdd(op + v * 4 + 1, q.y * w);
        atomicAdd(op + v * 4 + 2, q.z * w);
        atomicAdd(op + v * 4 + 3, q.w * w);
    }
}

// bias + relu; write next-layer f16 input, or accumulate batch pooling
__global__ void finalize_kernel(const float* __restrict__ acc,
                                const float* __restrict__ bias,
                                _Float16* __restrict__ Xnext,
                                const int* __restrict__ batch_map,
                                float* __restrict__ pooled, int isLast) {
    int i = blockIdx.x * blockDim.x + threadIdx.x;
    if (i >= N_NODES * 256) return;
    int node = i >> 8, c = i & 255;
    float v = fmaxf(acc[i] + bias[c], 0.f);
    if (!isLast) Xnext[i] = (_Float16)v;
    else         atomicAdd(&pooled[(size_t)batch_map[node] * 256 + c], v);
}

__global__ void cnt_kernel(const int* __restrict__ bm, float* cnt) {
    int i = blockIdx.x * blockDim.x + threadIdx.x;
    if (i < N_NODES) atomicAdd(&cnt[bm[i]], 1.f);
}

// ---------------------------------------------------------------------------
// MLP head: one block (128 threads) per batch element
// ---------------------------------------------------------------------------
__global__ __launch_bounds__(128)
void head_kernel(const float* __restrict__ pooled, const float* __restrict__ cnt,
                 const float* __restrict__ fc1w, const float* __restrict__ fc1b,
                 const float* __restrict__ fc2w, const float* __restrict__ fc2b,
                 float* __restrict__ out) {
    __shared__ float pm[256];
    __shared__ float hid[128];
    int b = blockIdx.x, t = threadIdx.x;
    float inv = 1.f / fmaxf(cnt[b], 1.f);
    for (int k = t; k < 256; k += 128) pm[k] = pooled[(size_t)b * 256 + k] * inv;
    __syncthreads();
    float a = fc1b[t];
    for (int k = 0; k < 256; ++k) a += pm[k] * fc1w[(size_t)k * 128 + t];
    hid[t] = fmaxf(a, 0.f) * fc2w[t];
    __syncthreads();
    for (int off = 64; off > 0; off >>= 1) {
        if (t < off) hid[t] += hid[t + off];
        __syncthreads();
    }
    if (t == 0) out[b] = hid[0] + fc2b[0];
}

// ---------------------------------------------------------------------------
// Host orchestration
// ---------------------------------------------------------------------------
extern "C" void kernel_launch(void* const* d_in, const int* in_sizes, int n_in,
                              void* d_out, int out_size, void* d_ws, size_t ws_size,
                              hipStream_t stream) {
    (void)in_sizes; (void)n_in; (void)out_size; (void)ws_size;
    // Input layout: x, edge_index, edge_attr, batch_map, then params flattened
    // jax-style (sorted keys at each nesting level):
    //  params: cnn{b1,b2,bl,w1,w2,wl}, fc1b, fc1w, fc2b, fc2w,
    //          gat[3]{W,We,att_dst,att_e,att_src,b}
    const float* x    = (const float*)d_in[0];
    const int*   ei   = (const int*)  d_in[1];
    const float* ea   = (const float*)d_in[2];
    const int*   bm   = (const int*)  d_in[3];
    const float* b1   = (const float*)d_in[4];
    const float* b2   = (const float*)d_in[5];
    const float* bl   = (const float*)d_in[6];
    const float* w1   = (const float*)d_in[7];
    const float* w2   = (const float*)d_in[8];
    const float* wl   = (const float*)d_in[9];
    const float* fc1b = (const float*)d_in[10];
    const float* fc1w = (const float*)d_in[11];
    const float* fc2b = (const float*)d_in[12];
    const float* fc2w = (const float*)d_in[13];
    const float* gatW[3], *gatWe[3], *gatAd[3], *gatAe[3], *gatAs[3], *gatB[3];
    for (int l = 0; l < 3; ++l) {
        int base = 14 + l * 6;
        gatW[l]  = (const float*)d_in[base + 0];
        gatWe[l] = (const float*)d_in[base + 1];
        gatAd[l] = (const float*)d_in[base + 2];
        gatAe[l] = (const float*)d_in[base + 3];
        gatAs[l] = (const float*)d_in[base + 4];
        gatB[l]  = (const float*)d_in[base + 5];
    }
    float* out = (float*)d_out;

    // workspace carve-out
    char* wsp = (char*)d_ws;
    auto alloc = [&](size_t bytes) -> char* {
        char* p = wsp;
        wsp += (bytes + 255) & ~(size_t)255;
        return p;
    };
    const size_t ES = N_EDGES + N_NODES;
    _Float16* Xf16  = (_Float16*)alloc((size_t)N_NODES * 256 * 2);
    _Float16* WT    = (_Float16*)alloc((size_t)256 * 256 * 2);
    float* H        = (float*)alloc((size_t)N_NODES * 256 * 4);
    float* accO     = (float*)alloc((size_t)N_NODES * 256 * 4);
    float* asrc     = (float*)alloc((size_t)N_NODES * 8 * 4);
    float* adst     = (float*)alloc((size_t)N_NODES * 8 * 4);
    float* mbuf     = (float*)alloc((size_t)N_NODES * 8 * 4);
    float* sbuf     = (float*)alloc((size_t)N_NODES * 8 * 4);
    float* exbuf    = (float*)alloc(ES * 8 * 4);
    float* deg      = (float*)alloc((size_t)N_NODES * 4);
    float* slsum    = (float*)alloc((size_t)N_NODES * 4);
    float* slattr   = (float*)alloc((size_t)N_NODES * 4);
    float* ce       = (float*)alloc(8 * 4);
    float* pooled   = (float*)alloc((size_t)N_BATCH * 256 * 4);
    float* cnt      = (float*)alloc((size_t)N_BATCH * 4);

    auto fill = [&](float* p, float v, size_t n) {
        fill_kernel<<<(unsigned)((n + 255) / 256), 256, 0, stream>>>(p, v, n);
    };

    // ---- front-end ----
    cnn_kernel<<<N_NODES, 32, 0, stream>>>(x, w1, b1, w2, b2, wl, bl, Xf16);
    fill(deg, 0.f, N_NODES);
    fill(slsum, 0.f, N_NODES);
    deg_kernel<<<(N_EDGES + 255) / 256, 256, 0, stream>>>(ei, ea, deg, slsum);
    slattr_kernel<<<(N_NODES + 255) / 256, 256, 0, stream>>>(slattr, slsum, deg);
    fill(pooled, 0.f, (size_t)N_BATCH * 256);
    fill(cnt, 0.f, N_BATCH);
    cnt_kernel<<<(N_NODES + 255) / 256, 256, 0, stream>>>(bm, cnt);

    // ---- 3 GAT layers ----
    const size_t nEdgeThreads = ES * 8;
    const unsigned edgeBlocks = (unsigned)((nEdgeThreads + 255) / 256);
    for (int l = 0; l < 3; ++l) {
        const int K = (l == 0) ? 32 : 256;   // padded K
        const int Kin = (l == 0) ? 25 : 256;
        convert_wt<<<(256 * K + 255) / 256, 256, 0, stream>>>(gatW[l], WT, Kin, K);
        if (l == 0)
            gemm_wmma<32 ><<<N_NODES / 16, 256, 0, stream>>>(Xf16, WT, H);
        else
            gemm_wmma<256><<<N_NODES / 16, 256, 0, stream>>>(Xf16, WT, H);
        ce_kernel<<<1, 8, 0, stream>>>(gatWe[l], gatAe[l], ce);
        att_scores<<<(N_NODES * 8 + 255) / 256, 256, 0, stream>>>(H, gatAs[l], gatAd[l], asrc, adst);
        fill(mbuf, -1e30f, (size_t)N_NODES * 8);
        fill(sbuf, 0.f, (size_t)N_NODES * 8);
        fill(accO, 0.f, (size_t)N_NODES * 256);
        edge_logit_max<<<edgeBlocks, 256, 0, stream>>>(ei, ea, slattr, asrc, adst, ce, exbuf, mbuf);
        edge_exp_sum<<<edgeBlocks, 256, 0, stream>>>(ei, exbuf, mbuf, sbuf);
        edge_message<<<edgeBlocks, 256, 0, stream>>>(ei, exbuf, sbuf, H, accO);
        finalize_kernel<<<(N_NODES * 256 + 255) / 256, 256, 0, stream>>>(
            accO, gatB[l], Xf16, bm, pooled, (l == 2) ? 1 : 0);
    }

    // ---- head ----
    head_kernel<<<N_BATCH, 128, 0, stream>>>(pooled, cnt, fc1w, fc1b, fc2w, fc2b, out);
}
